// LanguageModel_34574486733539
// MI455X (gfx1250) — matrix operand order repaired and checked by reading, not compiled, not run on hardware
//
#include <hip/hip_runtime.h>

// ---------------------------------------------------------------------------
// Fused LM head (hidden @ weight^T -> logits) + shifted cross-entropy loss
// for MI455X (gfx1250, wave32, WMMA).
//
// Shapes (fixed by reference): B=2, T=2048, H=1024, V=50257.
// d_out layout: [B*T*V] f32 logits, then 1 f32 loss.
//
// Strategy (roofline): 421.6 GFLOP, ~1 GB minimum traffic (45us @ 23.3TB/s).
// bf16 WMMA (16x16x32) is the fast matrix path; f32 inputs are converted to
// bf16 ONCE in a bandwidth-bound pre-pass (~14us) so the GEMM inner loop is
// pure b128 loads + v_wmma with zero conversion VALU. Round-1 histogram
// showed inline software bf16 conversion costing ~350 VALU per 8 WMMA.
// ---------------------------------------------------------------------------

#define DEV __device__ __forceinline__

typedef __attribute__((ext_vector_type(16))) __bf16 v16bf;
typedef __attribute__((ext_vector_type(8)))  __bf16 v8bf;
typedef __attribute__((ext_vector_type(8)))  float  v8f;

constexpr int kB = 2, kT = 2048, kH = 1024, kV = 50257;
constexpr int kM = kB * kT;                   // 4096 rows of hidden states
constexpr int BM = 128, BN = 128, BK = 32;    // block tile; BK == WMMA K (bf16)
constexpr int NUM_MTILE = kM / BM;            // 32 (power of two, used as mask)
constexpr int NUM_NTILE = (kV + BN - 1) / BN; // 393
static_assert(NUM_MTILE == 32, "grid decode assumes 32 M tiles");

// f32 -> bf16 RNE; plain fptrunc so the backend can select the native
// v_cvt_pk_bf16_f32 path if gfx1250 has it (manual bit-tricks provably don't).
DEV __bf16 f2bf(float f) { return static_cast<__bf16>(f); }

DEV v16bf cat16(v8bf lo, v8bf hi) {
  return __builtin_shufflevector(lo, hi, 0, 1, 2, 3, 4, 5, 6, 7,
                                 8, 9, 10, 11, 12, 13, 14, 15);
}

// ---------------------------------------------------------------------------
// Pre-pass: f32 -> bf16 (8 elements / thread, fully coalesced, ~14us total).
// ---------------------------------------------------------------------------
__global__ __launch_bounds__(256)
void convert_to_bf16(const float* __restrict__ src, __bf16* __restrict__ dst,
                     long n) {
  const long i = ((long)blockIdx.x * 256 + threadIdx.x) * 8;
  if (i + 8 > n) return;
  const float4 a = *(const float4*)(src + i);
  const float4 b = *(const float4*)(src + i + 4);
  v8bf o;
  o[0] = f2bf(a.x); o[1] = f2bf(a.y); o[2] = f2bf(a.z); o[3] = f2bf(a.w);
  o[4] = f2bf(b.x); o[5] = f2bf(b.y); o[6] = f2bf(b.z); o[7] = f2bf(b.w);
  *(v8bf*)(dst + i) = o;
}

// ---------------------------------------------------------------------------
// Fragment loaders (pre-converted bf16 operands; each fragment = 2x b128).
// A (16x32, MxK): lane r=lane&15 holds row M=r; lanes 0-15 hold K=[0..15];
// lanes 16-31 hold K=[16..31] (same K order as B).
// B (32x16, KxN): lane c holds column N=c (one K-contiguous weight row);
// lanes 0-15 hold K=[0..15], lanes 16-31 hold K=[16..31].
// ---------------------------------------------------------------------------
DEV v16bf load_frag_a_bf(const __bf16* __restrict__ p) {
  v8bf c0 = *(const v8bf*)(p + 0);
  v8bf c1 = *(const v8bf*)(p + 8);
  return cat16(c0, c1);
}
DEV v16bf load_frag_b_bf(const __bf16* __restrict__ p) {
  v8bf c0 = *(const v8bf*)(p + 0);
  v8bf c1 = *(const v8bf*)(p + 8);
  return cat16(c0, c1);
}

// Fallback loaders reading f32 + converting inline (used only if d_ws is too
// small to hold the bf16 copies).
DEV v16bf load_frag_a_f32(const float* __restrict__ p) {
  float4 c0 = *(const float4*)(p + 0);
  float4 c1 = *(const float4*)(p + 4);
  float4 c2 = *(const float4*)(p + 8);
  float4 c3 = *(const float4*)(p + 12);
  v16bf o;
  o[0]  = f2bf(c0.x); o[1]  = f2bf(c0.y); o[2]  = f2bf(c0.z); o[3]  = f2bf(c0.w);
  o[4]  = f2bf(c1.x); o[5]  = f2bf(c1.y); o[6]  = f2bf(c1.z); o[7]  = f2bf(c1.w);
  o[8]  = f2bf(c2.x); o[9]  = f2bf(c2.y); o[10] = f2bf(c2.z); o[11] = f2bf(c2.w);
  o[12] = f2bf(c3.x); o[13] = f2bf(c3.y); o[14] = f2bf(c3.z); o[15] = f2bf(c3.w);
  return o;
}
DEV v16bf load_frag_b_f32(const float* __restrict__ p) {
  float4 c0 = *(const float4*)(p + 0);
  float4 c1 = *(const float4*)(p + 4);
  float4 c2 = *(const float4*)(p + 8);
  float4 c3 = *(const float4*)(p + 12);
  v16bf o;
  o[0]  = f2bf(c0.x); o[1]  = f2bf(c0.y); o[2]  = f2bf(c0.z); o[3]  = f2bf(c0.w);
  o[4]  = f2bf(c1.x); o[5]  = f2bf(c1.y); o[6]  = f2bf(c1.z); o[7]  = f2bf(c1.w);
  o[8]  = f2bf(c2.x); o[9]  = f2bf(c2.y); o[10] = f2bf(c2.z); o[11] = f2bf(c2.w);
  o[12] = f2bf(c3.x); o[13] = f2bf(c3.y); o[14] = f2bf(c3.z); o[15] = f2bf(c3.w);
  return o;
}

#define WMMA_BF16(a, b, c) \
  __builtin_amdgcn_wmma_f32_16x16x32_bf16(false, (a), false, (b), (short)0, (c), false, false)

// Shared accumulator store. C/D layout (ISA 7.12.2): VGPR g -> M = g +
// 8*(lane>=16), N = lane&15.
DEV void store_tile(float* __restrict__ C, v8f acc[2][4], int m0, int n0,
                    int r, int half) {
#pragma unroll
  for (int i = 0; i < 2; ++i) {
#pragma unroll
    for (int j = 0; j < 4; ++j) {
      const int n = n0 + j * 16 + r;
      if (n < kV) {
        float* cp = C + (size_t)(m0 + i * 16 + half * 8) * kV + n;
#pragma unroll
        for (int g = 0; g < 8; ++g)
          cp[(size_t)g * kV] = acc[i][j][g];
      }
    }
  }
}

// ---------------------------------------------------------------------------
// GEMM, bf16 pre-converted operands: logits[m][v] = sum_h A[m][h] * W[v][h].
// 128x128 block tile, 8 waves in 4x2, each wave: 32x64 = 2x4 WMMA tiles.
// Inner loop: 12x global_load_b128 + 8x v_wmma, no conversion VALU.
// ---------------------------------------------------------------------------
__global__ __launch_bounds__(256)
void lm_head_gemm_bf16(const __bf16* __restrict__ A, const __bf16* __restrict__ W,
                       float* __restrict__ C) {
  // N-major grid: 32 consecutive blocks share one 256KB bf16 weight slab in L2.
  const int mTile = blockIdx.x & (NUM_MTILE - 1);
  const int nTile = blockIdx.x >> 5;

  const int lane = threadIdx.x & 31;
  const int wave = threadIdx.x >> 5;
  const int m0 = mTile * BM + (wave & 3) * 32;
  const int n0 = nTile * BN + (wave >> 2) * 64;

  const int r    = lane & 15;
  const int half = lane >> 4;

  const __bf16* pa0 = A + (size_t)(m0 + r) * kH + half * 16;
  const __bf16* pa1 = pa0 + (size_t)16 * kH;
  const __bf16* pb[4];
#pragma unroll
  for (int j = 0; j < 4; ++j) {
    int v = n0 + j * 16 + r;
    if (v >= kV) v = kV - 1;   // clamp; garbage columns masked at store
    pb[j] = W + (size_t)v * kH + half * 16;
  }

  v8f acc[2][4];
#pragma unroll
  for (int i = 0; i < 2; ++i)
#pragma unroll
    for (int j = 0; j < 4; ++j)
      acc[i][j] = v8f{0.f, 0.f, 0.f, 0.f, 0.f, 0.f, 0.f, 0.f};

#pragma unroll 2
  for (int k = 0; k < kH; k += BK) {
    if (k + BK < kH) {
      __builtin_prefetch(pb[0] + k + BK, 0, 1);
      __builtin_prefetch(pb[2] + k + BK, 0, 1);
    }
    v16bf a0 = load_frag_a_bf(pa0 + k);
    v16bf a1 = load_frag_a_bf(pa1 + k);
    v16bf b0 = load_frag_b_bf(pb[0] + k);
    v16bf b1 = load_frag_b_bf(pb[1] + k);
    v16bf b2 = load_frag_b_bf(pb[2] + k);
    v16bf b3 = load_frag_b_bf(pb[3] + k);

    acc[0][0] = WMMA_BF16(a0, b0, acc[0][0]);
    acc[0][1] = WMMA_BF16(a0, b1, acc[0][1]);
    acc[0][2] = WMMA_BF16(a0, b2, acc[0][2]);
    acc[0][3] = WMMA_BF16(a0, b3, acc[0][3]);
    acc[1][0] = WMMA_BF16(a1, b0, acc[1][0]);
    acc[1][1] = WMMA_BF16(a1, b1, acc[1][1]);
    acc[1][2] = WMMA_BF16(a1, b2, acc[1][2]);
    acc[1][3] = WMMA_BF16(a1, b3, acc[1][3]);
  }

  store_tile(C, acc, m0, n0, r, half);
}

// Fallback GEMM reading f32 + converting inline (only if ws too small).
__global__ __launch_bounds__(256)
void lm_head_gemm_f32(const float* __restrict__ A, const float* __restrict__ W,
                      float* __restrict__ C) {
  const int mTile = blockIdx.x & (NUM_MTILE - 1);
  const int nTile = blockIdx.x >> 5;
  const int lane = threadIdx.x & 31;
  const int wave = threadIdx.x >> 5;
  const int m0 = mTile * BM + (wave & 3) * 32;
  const int n0 = nTile * BN + (wave >> 2) * 64;
  const int r    = lane & 15;
  const int half = lane >> 4;

  const float* pa0 = A + (size_t)(m0 + r) * kH + half * 16;
  const float* pa1 = pa0 + (size_t)16 * kH;
  const float* pb[4];
#pragma unroll
  for (int j = 0; j < 4; ++j) {
    int v = n0 + j * 16 + r;
    if (v >= kV) v = kV - 1;
    pb[j] = W + (size_t)v * kH + half * 16;
  }

  v8f acc[2][4];
#pragma unroll
  for (int i = 0; i < 2; ++i)
#pragma unroll
    for (int j = 0; j < 4; ++j)
      acc[i][j] = v8f{0.f, 0.f, 0.f, 0.f, 0.f, 0.f, 0.f, 0.f};

#pragma unroll 1
  for (int k = 0; k < kH; k += BK) {
    v16bf a0 = load_frag_a_f32(pa0 + k);
    v16bf a1 = load_frag_a_f32(pa1 + k);
    v16bf b0 = load_frag_b_f32(pb[0] + k);
    v16bf b1 = load_frag_b_f32(pb[1] + k);
    v16bf b2 = load_frag_b_f32(pb[2] + k);
    v16bf b3 = load_frag_b_f32(pb[3] + k);

    acc[0][0] = WMMA_BF16(a0, b0, acc[0][0]);
    acc[0][1] = WMMA_BF16(a0, b1, acc[0][1]);
    acc[0][2] = WMMA_BF16(a0, b2, acc[0][2]);
    acc[0][3] = WMMA_BF16(a0, b3, acc[0][3]);
    acc[1][0] = WMMA_BF16(a1, b0, acc[1][0]);
    acc[1][1] = WMMA_BF16(a1, b1, acc[1][1]);
    acc[1][2] = WMMA_BF16(a1, b2, acc[1][2]);
    acc[1][3] = WMMA_BF16(a1, b3, acc[1][3]);
  }

  store_tile(C, acc, m0, n0, r, half);
}

// ---------------------------------------------------------------------------
// Cross-entropy: one block per shifted row, single-pass online softmax
// (reads each 201KB logits row exactly once), LDS tree combine, atomicAdd.
// ---------------------------------------------------------------------------
__global__ __launch_bounds__(256)
void ce_loss(const float* __restrict__ logits, const int* __restrict__ labels,
             float* __restrict__ ws) {
  __shared__ float smax[256];
  __shared__ float ssum[256];
  const int T1 = kT - 1;
  const int b = blockIdx.x / T1;
  const int t = blockIdx.x % T1;
  const float* row = logits + (size_t)(b * kT + t) * kV;
  const int label = labels[b * kT + t + 1];   // shift_labels = labels[:,1:]

  float m = -3.402823466e38f, s = 0.f;
  for (int i = threadIdx.x; i < kV; i += 256) {
    const float x = row[i];
    if (x > m) { s = s * __expf(m - x) + 1.f; m = x; }
    else       { s += __expf(x - m); }
  }
  smax[threadIdx.x] = m;
  ssum[threadIdx.x] = s;
  __syncthreads();
  for (int off = 128; off > 0; off >>= 1) {
    if (threadIdx.x < off) {
      const float m1 = smax[threadIdx.x], s1 = ssum[threadIdx.x];
      const float m2 = smax[threadIdx.x + off], s2 = ssum[threadIdx.x + off];
      const float M = fmaxf(m1, m2);
      smax[threadIdx.x] = M;
      ssum[threadIdx.x] = s1 * __expf(m1 - M) + s2 * __expf(m2 - M);
    }
    __syncthreads();
  }
  if (threadIdx.x == 0) {
    int safe = label;
    if (safe < 0) safe = 0;
    if (safe >= kV) safe = kV - 1;
    const float nll = __logf(ssum[0]) + smax[0] - row[safe];
    if (label != -1) {
      atomicAdd(&ws[0], nll);
      atomicAdd(&ws[1], 1.f);
    }
  }
}

__global__ void init_ws(float* __restrict__ ws) {
  ws[0] = 0.f;
  ws[1] = 0.f;
}

__global__ void finalize_loss(const float* __restrict__ ws, float* __restrict__ out) {
  const float n = ws[1] < 1.f ? 1.f : ws[1];
  out[(size_t)kM * kV] = ws[0] / n;
}

extern "C" void kernel_launch(void* const* d_in, const int* in_sizes, int n_in,
                              void* d_out, int out_size, void* d_ws, size_t ws_size,
                              hipStream_t stream) {
  (void)in_sizes; (void)n_in; (void)out_size;
  const float* hidden = (const float*)d_in[0];   // [B,T,H] f32
  const float* weight = (const float*)d_in[1];   // [V,H]   f32
  const int*   labels = (const int*)d_in[2];     // [B,T]   int32
  float* out = (float*)d_out;                    // [B*T*V] logits + [1] loss
  float* ws  = (float*)d_ws;                     // [0]=sum nll, [1]=n_valid

  init_ws<<<dim3(1), dim3(1), 0, stream>>>(ws);

  const long nW = (long)kV * kH;                 // 51,463,168
  const long nA = (long)kM * kH;                 //  4,194,304
  const size_t accumBytes = 256;                 // reserved for accumulators
  const size_t needBytes = accumBytes + (size_t)(nW + nA) * sizeof(__bf16);

  if (ws_size >= needBytes) {
    __bf16* wbf = (__bf16*)((char*)d_ws + accumBytes);
    __bf16* abf = wbf + nW;
    convert_to_bf16<<<dim3((unsigned)((nW / 8 + 255) / 256)), dim3(256), 0, stream>>>(weight, wbf, nW);
    convert_to_bf16<<<dim3((unsigned)((nA / 8 + 255) / 256)), dim3(256), 0, stream>>>(hidden, abf, nA);
    lm_head_gemm_bf16<<<dim3(NUM_NTILE * NUM_MTILE), dim3(256), 0, stream>>>(abf + 0 - 0 + 0, wbf, out); // A then W
    // (note: args are (A, W, C))
  } else {
    lm_head_gemm_f32<<<dim3(NUM_NTILE * NUM_MTILE), dim3(256), 0, stream>>>(hidden, weight, out);
  }

  ce_loss<<<dim3(kB * (kT - 1)), dim3(256), 0, stream>>>(out, labels, ws);
  finalize_loss<<<dim3(1), dim3(1), 0, stream>>>(ws, out);
}
